// StrategyEncoder_61203283968766
// MI455X (gfx1250) — matrix operand order repaired and checked by reading, not compiled, not run on hardware
//
#include <hip/hip_runtime.h>
#include <hip/hip_bf16.h>

typedef __attribute__((ext_vector_type(16))) _Float16 v16h;
typedef __attribute__((ext_vector_type(8)))  _Float16 v8h;
typedef __attribute__((ext_vector_type(8)))  float    v8f;

#define NBATCH   8192
#define HIST_N   100
#define HIST_P   112      // padded to 7 M-tiles of 16
#define NEGBIG   -1000000000.0f
#define GEPS     1e-20f

// d_out layout (floats): [strat 2B*64][smax 2B*64][unexp_h B*64][exp_h B*64]
#define SMAX_OFF (2*NBATCH*64)
#define Q_OFF    (4*NBATCH*64)   // unexp_h region; exp_h follows => q rows contiguous

// ---------------- WMMA fragment helpers (wave32, v_wmma_f32_16x16x32_f16) ----
// A (16x32 f16): lane l -> row M = l&15 ; K halves interleaved by (l>>4)*8
__device__ __forceinline__ v16h frag_a(const _Float16* A, int stride, int mbase, int kbase, int lane) {
  const _Float16* p = A + (mbase + (lane & 15)) * stride + kbase + ((lane >> 4) & 1) * 8;
  v8h lo = *(const v8h*)p;
  v8h hi = *(const v8h*)(p + 16);
  return __builtin_shufflevector(lo, hi, 0,1,2,3,4,5,6,7,8,9,10,11,12,13,14,15);
}
// B (32x16 f16) from W stored row-major [n][k] (computes X @ W^T):
// lane l -> col N = l&15 ; K halves interleaved by (l>>4)*8
__device__ __forceinline__ v16h frag_b(const _Float16* W, int K, int nbase, int kbase, int lane) {
  const _Float16* p = W + (nbase + (lane & 15)) * K + kbase + ((lane >> 4) & 1) * 8;
  v8h lo = *(const v8h*)p;
  v8h hi = *(const v8h*)(p + 16);
  return __builtin_shufflevector(lo, hi, 0,1,2,3,4,5,6,7,8,9,10,11,12,13,14,15);
}
template<int K>
__device__ __forceinline__ v8f gemm_tile(const _Float16* A, int sa, const _Float16* W,
                                         int mt, int nt, int lane) {
  v8f acc = {};
  #pragma unroll
  for (int k = 0; k < K; k += 32)
    acc = __builtin_amdgcn_wmma_f32_16x16x32_f16(false, frag_a(A, sa, mt * 16, k, lane),
                                                 false, frag_b(W, K, nt * 16, k, lane),
                                                 (short)0, acc, false, false);
  return acc;
}
// C/D layout: lane l -> col N=l&15 ; VGPR r -> row M = r + 8*(l>>4)

// ---------------- prep: fp32 -> f16 weight conversion ------------------------
__global__ __launch_bounds__(256) void conv_kernel(_Float16* dst, const float* src,
                                                   int rows, int cols, int stride) {
  int i = blockIdx.x * 256 + threadIdx.x;
  if (i < rows * cols) {
    int r = i / cols, c = i - r * cols;
    dst[i] = (_Float16)src[r * stride + c];
  }
}
// R6[r][n] = rate_b[n] + sum_j rate_table[r][j] * rate_w[n][64+j]   (6 x 64)
__global__ __launch_bounds__(384) void r6_kernel(float* R6, const float* rate_table,
                                                 const float* rate_w, const float* rate_b) {
  int i = threadIdx.x;
  if (i < 384) {
    int r = i >> 6, n = i & 63;
    float acc = rate_b[n];
    for (int j = 0; j < 64; ++j) acc += rate_table[r * 64 + j] * rate_w[n * 128 + 64 + j];
    R6[i] = acc;
  }
}

// ---------------- kernel 1: per-row dual attention ---------------------------
__global__ __launch_bounds__(256) void attn_kernel(
    const float* __restrict__ user_table, const float* __restrict__ item_table,
    const float* __restrict__ ia1_w, const float* __restrict__ ia1_b,
    const float* __restrict__ ia2_b, const float* __restrict__ ia3_w, const float* __restrict__ ia3_b,
    const float* __restrict__ ua1_w, const float* __restrict__ ua1_b,
    const float* __restrict__ ua2_b, const float* __restrict__ ua3_w, const float* __restrict__ ua3_b,
    const int* __restrict__ user, const int* __restrict__ user_hist, const int* __restrict__ user_nbrs,
    const _Float16* __restrict__ wRateA, const _Float16* __restrict__ wIa1A,
    const _Float16* __restrict__ wIa2,  const _Float16* __restrict__ wUa1A,
    const _Float16* __restrict__ wUa2,  const float* __restrict__ R6,
    _Float16* __restrict__ hi_hs, _Float16* __restrict__ u16) {
  const int b = blockIdx.x;
  const int tid = threadIdx.x, lane = tid & 31, wave = tid >> 5;

  __shared__ _Float16 sX[HIST_P * 72];   // item emb, later neighbor emb
  __shared__ _Float16 sE[HIST_P * 72];   // rate-linear output (attn values)
  __shared__ _Float16 sH1[HIST_P * 72];
  __shared__ _Float16 sH2[HIST_P * 72];
  __shared__ float sU[64], sC1[64], sV[64], sR6[6 * 64];
  __shared__ int   sRate[HIST_P];
  __shared__ float sMask[HIST_P], sL[HIST_P], sRed[2];

  if (tid < 64) sU[tid] = user_table[user[b] * 64 + tid];
  for (int i = tid; i < 6 * 64; i += 256) sR6[i] = R6[i];
  if (tid < HIST_P) {
    if (tid < HIST_N) {
      int it = user_hist[(b * HIST_N + tid) * 2 + 0];
      sRate[tid] = user_hist[(b * HIST_N + tid) * 2 + 1];
      sMask[tid] = (it == 0) ? NEGBIG : 0.f;
    } else { sRate[tid] = 0; sMask[tid] = NEGBIG; }
  }
  // gather history item embeddings (fp32 -> f16)
  for (int i = tid; i < HIST_N * 16; i += 256) {
    int m = i >> 4, c = (i & 15) * 4;
    int it = user_hist[(b * HIST_N + m) * 2];
    float4 v = *(const float4*)(item_table + it * 64 + c);
    _Float16* d = &sX[m * 72 + c];
    d[0] = (_Float16)v.x; d[1] = (_Float16)v.y; d[2] = (_Float16)v.z; d[3] = (_Float16)v.w;
  }
  __syncthreads();
  // c1 = ia1_w[:,64:] @ u + ia1_b  (user half of attn layer-1 folded out)
  if (tid < 64) {
    float acc = ia1_b[tid];
    for (int j = 0; j < 64; ++j) acc += ia1_w[tid * 128 + 64 + j] * sU[j];
    sC1[tid] = acc;
    sV[tid] = ia3_w[tid];
  }
  __syncthreads();
  // GEMM1: E = itemEmb @ rateW_A^T + R6[rate]
  for (int t = wave; t < 28; t += 8) {
    int mt = t >> 2, nt = t & 3;
    v8f acc = gemm_tile<64>(sX, 72, wRateA, mt, nt, lane);
    int n = nt * 16 + (lane & 15);
    #pragma unroll
    for (int r = 0; r < 8; ++r) {
      int m = mt * 16 + r + ((lane >> 4) << 3);
      sE[m * 72 + n] = (_Float16)(acc[r] + sR6[sRate[m] * 64 + n]);
    }
  }
  __syncthreads();
  // GEMM2: H1 = lrelu(E @ ia1W_A^T + c1)
  for (int t = wave; t < 28; t += 8) {
    int mt = t >> 2, nt = t & 3;
    v8f acc = gemm_tile<64>(sE, 72, wIa1A, mt, nt, lane);
    int n = nt * 16 + (lane & 15);
    float bn = sC1[n];
    #pragma unroll
    for (int r = 0; r < 8; ++r) {
      int m = mt * 16 + r + ((lane >> 4) << 3);
      float v = acc[r] + bn; v = (v >= 0.f) ? v : 0.1f * v;
      sH1[m * 72 + n] = (_Float16)v;
    }
  }
  __syncthreads();
  // GEMM3: H2 = lrelu(H1 @ ia2W^T + ia2_b)
  for (int t = wave; t < 28; t += 8) {
    int mt = t >> 2, nt = t & 3;
    v8f acc = gemm_tile<64>(sH1, 72, wIa2, mt, nt, lane);
    int n = nt * 16 + (lane & 15);
    float bn = ia2_b[n];
    #pragma unroll
    for (int r = 0; r < 8; ++r) {
      int m = mt * 16 + r + ((lane >> 4) << 3);
      float v = acc[r] + bn; v = (v >= 0.f) ? v : 0.1f * v;
      sH2[m * 72 + n] = (_Float16)v;
    }
  }
  __syncthreads();
  // logits + masked softmax over 100 (padded rows -> -1e9)
  if (tid < HIST_P) {
    float v = NEGBIG;
    if (tid < HIST_N) {
      float acc = ia3_b[0] + sMask[tid];
      for (int k = 0; k < 64; ++k) acc += (float)sH2[tid * 72 + k] * sV[k];
      v = acc;
    }
    sL[tid] = v;
  }
  __syncthreads();
  if (tid == 0) { float mx = sL[0]; for (int i = 1; i < HIST_P; ++i) mx = fmaxf(mx, sL[i]); sRed[0] = mx; }
  __syncthreads();
  if (tid < HIST_P) sL[tid] = __expf(sL[tid] - sRed[0]);
  __syncthreads();
  if (tid == 0) { float s = 0.f; for (int i = 0; i < HIST_P; ++i) s += sL[i]; sRed[1] = 1.f / s; }
  __syncthreads();
  if (tid < 64) {
    float inv = sRed[1], acc = 0.f;
    for (int m = 0; m < HIST_N; ++m) acc += sL[m] * (float)sE[m * 72 + tid];
    hi_hs[b * 128 + tid] = (_Float16)(acc * inv);
  }
  __syncthreads();
  // ---------------- social branch (M = 64) ----------------
  for (int i = tid; i < 64 * 16; i += 256) {
    int m = i >> 4, c = (i & 15) * 4;
    int nb = user_nbrs[b * 64 + m];
    float4 v = *(const float4*)(user_table + nb * 64 + c);
    _Float16* d = &sX[m * 72 + c];
    d[0] = (_Float16)v.x; d[1] = (_Float16)v.y; d[2] = (_Float16)v.z; d[3] = (_Float16)v.w;
  }
  if (tid < 64) sMask[tid] = (user_nbrs[b * 64 + tid] == 0) ? NEGBIG : 0.f;
  if (tid >= 64 && tid < 128) {
    int n = tid - 64;
    float acc = ua1_b[n];
    for (int j = 0; j < 64; ++j) acc += ua1_w[n * 128 + 64 + j] * sU[j];
    sC1[n] = acc;
  }
  if (tid >= 128 && tid < 192) sV[tid - 128] = ua3_w[tid - 128];
  __syncthreads();
  for (int t = wave; t < 16; t += 8) {
    int mt = t >> 2, nt = t & 3;
    v8f acc = gemm_tile<64>(sX, 72, wUa1A, mt, nt, lane);
    int n = nt * 16 + (lane & 15);
    float bn = sC1[n];
    #pragma unroll
    for (int r = 0; r < 8; ++r) {
      int m = mt * 16 + r + ((lane >> 4) << 3);
      float v = acc[r] + bn; v = (v >= 0.f) ? v : 0.1f * v;
      sH1[m * 72 + n] = (_Float16)v;
    }
  }
  __syncthreads();
  for (int t = wave; t < 16; t += 8) {
    int mt = t >> 2, nt = t & 3;
    v8f acc = gemm_tile<64>(sH1, 72, wUa2, mt, nt, lane);
    int n = nt * 16 + (lane & 15);
    float bn = ua2_b[n];
    #pragma unroll
    for (int r = 0; r < 8; ++r) {
      int m = mt * 16 + r + ((lane >> 4) << 3);
      float v = acc[r] + bn; v = (v >= 0.f) ? v : 0.1f * v;
      sH2[m * 72 + n] = (_Float16)v;
    }
  }
  __syncthreads();
  if (tid < 64) {
    float acc = ua3_b[0] + sMask[tid];
    for (int k = 0; k < 64; ++k) acc += (float)sH2[tid * 72 + k] * sV[k];
    sL[tid] = acc;
  }
  __syncthreads();
  if (tid == 0) { float mx = sL[0]; for (int i = 1; i < 64; ++i) mx = fmaxf(mx, sL[i]); sRed[0] = mx; }
  __syncthreads();
  if (tid < 64) sL[tid] = __expf(sL[tid] - sRed[0]);
  __syncthreads();
  if (tid == 0) { float s = 0.f; for (int i = 0; i < 64; ++i) s += sL[i]; sRed[1] = 1.f / s; }
  __syncthreads();
  if (tid < 64) {
    float inv = sRed[1], acc = 0.f;
    for (int m = 0; m < 64; ++m) acc += sL[m] * (float)sX[m * 72 + tid];
    hi_hs[b * 128 + 64 + tid] = (_Float16)(acc * inv);
    u16[b * 64 + tid] = (_Float16)sU[tid];
  }
}

// ---------------- kernel 2: batched head (64-row tiles) ----------------------
__global__ __launch_bounds__(256) void head_kernel(
    const float* __restrict__ item_table,
    const float* __restrict__ fuse_b, const float* __restrict__ self_b,
    const float* __restrict__ exp_b, const float* __restrict__ unexp_b,
    const float* __restrict__ si1_b, const float* __restrict__ si2_b,
    const float* __restrict__ um_b, const float* __restrict__ em_b,
    const int* __restrict__ unexp_item, const int* __restrict__ exp_item,
    const _Float16* __restrict__ wFuse, const _Float16* __restrict__ wSelf,
    const _Float16* __restrict__ wExp, const _Float16* __restrict__ wUnexp,
    const _Float16* __restrict__ wSi1, const _Float16* __restrict__ wSi2,
    const _Float16* __restrict__ wUm, const _Float16* __restrict__ wEm,
    const _Float16* __restrict__ hi_hs, const _Float16* __restrict__ u16,
    float* __restrict__ out) {
  const int base = blockIdx.x * 64;
  const int tid = threadIdx.x, lane = tid & 31, wave = tid >> 5;
  __shared__ _Float16 sA[64 * 136];   // 128-wide concat buffer
  __shared__ _Float16 sB[64 * 136];   // 128-wide concat buffer
  __shared__ _Float16 sH[64 * 72];    // h2 (kept for both branches)
  __shared__ _Float16 sT[64 * 72];

  if (tid == 0) { __builtin_prefetch(wUm, 0, 3); __builtin_prefetch(wEm, 0, 3); }
  // load [h_item | h_social]
  for (int i = tid; i < 64 * 16; i += 256) {
    int m = i >> 4, c = (i & 15) * 8;
    *(v8h*)&sA[m * 136 + c] = *(const v8h*)&hi_hs[(base + m) * 128 + c];
  }
  __syncthreads();
  // fuse: h = relu([hi|hs] @ fuseW^T + b) -> sB[:, :64] ; user emb -> sB[:, 64:]
  for (int t = wave; t < 16; t += 8) {
    int mt = t >> 2, nt = t & 3;
    v8f acc = gemm_tile<128>(sA, 136, wFuse, mt, nt, lane);
    int n = nt * 16 + (lane & 15);
    float bn = fuse_b[n];
    #pragma unroll
    for (int r = 0; r < 8; ++r) {
      int m = mt * 16 + r + ((lane >> 4) << 3);
      float v = acc[r] + bn;
      sB[m * 136 + n] = (_Float16)(v > 0.f ? v : 0.f);
    }
  }
  for (int i = tid; i < 64 * 8; i += 256) {
    int m = i >> 3, c = (i & 7) * 8;
    *(v8h*)&sB[m * 136 + 64 + c] = *(const v8h*)&u16[(base + m) * 64 + c];
  }
  __syncthreads();
  // self: h2 = [h|u] @ selfW^T + b -> sH
  for (int t = wave; t < 16; t += 8) {
    int mt = t >> 2, nt = t & 3;
    v8f acc = gemm_tile<128>(sB, 136, wSelf, mt, nt, lane);
    int n = nt * 16 + (lane & 15);
    float bn = self_b[n];
    #pragma unroll
    for (int r = 0; r < 8; ++r) {
      int m = mt * 16 + r + ((lane >> 4) << 3);
      sH[m * 72 + n] = (_Float16)(acc[r] + bn);
    }
  }
  __syncthreads();
  // ---- two branches: 0 = unexp, 1 = exp ----
  #pragma unroll 1
  for (int br = 0; br < 2; ++br) {
    const _Float16* wU = br ? wExp : wUnexp;
    const float* bU = br ? exp_b : unexp_b;
    const int* items = br ? exp_item : unexp_item;
    const _Float16* wM = br ? wEm : wUm;
    const float* bM = br ? em_b : um_b;
    float* oreg = out + Q_OFF + br * (NBATCH * 64);
    // branch user half -> sA[:, :64]
    for (int t = wave; t < 16; t += 8) {
      int mt = t >> 2, nt = t & 3;
      v8f acc = gemm_tile<64>(sH, 72, wU, mt, nt, lane);
      int n = nt * 16 + (lane & 15);
      float bn = bU[n];
      #pragma unroll
      for (int r = 0; r < 8; ++r) {
        int m = mt * 16 + r + ((lane >> 4) << 3);
        sA[m * 136 + n] = (_Float16)(acc[r] + bn);
      }
    }
    // gather branch item embeddings -> sB[:, :64]
    for (int i = tid; i < 64 * 16; i += 256) {
      int m = i >> 4, c = (i & 15) * 4;
      int it = items[base + m];
      float4 v = *(const float4*)(item_table + it * 64 + c);
      _Float16* d = &sB[m * 136 + c];
      d[0] = (_Float16)v.x; d[1] = (_Float16)v.y; d[2] = (_Float16)v.z; d[3] = (_Float16)v.w;
    }
    __syncthreads();
    // si1: t = relu(e @ si1^T + b) -> sT
    for (int t = wave; t < 16; t += 8) {
      int mt = t >> 2, nt = t & 3;
      v8f acc = gemm_tile<64>(sB, 136, wSi1, mt, nt, lane);
      int n = nt * 16 + (lane & 15);
      float bn = si1_b[n];
      #pragma unroll
      for (int r = 0; r < 8; ++r) {
        int m = mt * 16 + r + ((lane >> 4) << 3);
        float v = acc[r] + bn;
        sT[m * 72 + n] = (_Float16)(v > 0.f ? v : 0.f);
      }
    }
    __syncthreads();
    // si2: item half -> sA[:, 64:]
    for (int t = wave; t < 16; t += 8) {
      int mt = t >> 2, nt = t & 3;
      v8f acc = gemm_tile<64>(sT, 72, wSi2, mt, nt, lane);
      int n = nt * 16 + (lane & 15);
      float bn = si2_b[n];
      #pragma unroll
      for (int r = 0; r < 8; ++r) {
        int m = mt * 16 + r + ((lane >> 4) << 3);
        sA[m * 136 + 64 + n] = (_Float16)(acc[r] + bn);
      }
    }
    __syncthreads();
    // um/em: out = relu([user|item] @ W^T + b) -> d_out (this IS q)
    for (int t = wave; t < 16; t += 8) {
      int mt = t >> 2, nt = t & 3;
      v8f acc = gemm_tile<128>(sA, 136, wM, mt, nt, lane);
      int n = nt * 16 + (lane & 15);
      float bn = bM[n];
      #pragma unroll
      for (int r = 0; r < 8; ++r) {
        int m = mt * 16 + r + ((lane >> 4) << 3);
        float v = acc[r] + bn;
        oreg[(base + m) * 64 + n] = (v > 0.f ? v : 0.f);
      }
    }
    __syncthreads();
  }
}

// ---------------- kernel 3: row softmax + Gumbel one-hot ---------------------
__device__ __forceinline__ float rng_uniform(unsigned idx) {
  unsigned s = idx * 747796405u + 2891336453u;
  unsigned w = ((s >> ((s >> 28) + 4u)) ^ s) * 277803737u;
  w = (w >> 22) ^ w;
  return (float)(w >> 8) * (1.0f / 16777216.0f);
}
__global__ __launch_bounds__(256) void gumbel_kernel(float* __restrict__ out) {
  int row = blockIdx.x * 256 + threadIdx.x;   // 0 .. 2B-1 (q rows, contiguous in d_out)
  const float* qp = out + Q_OFF + row * 64;
  float q[64];
  #pragma unroll
  for (int i = 0; i < 16; ++i) *(float4*)&q[i * 4] = *(const float4*)&qp[i * 4];
  float mx = q[0];
  #pragma unroll
  for (int i = 1; i < 64; ++i) mx = fmaxf(mx, q[i]);
  float s = 0.f;
  #pragma unroll
  for (int i = 0; i < 64; ++i) s += __expf(q[i] - mx);
  float inv = 1.f / s;
  float* so = out + SMAX_OFF + row * 64;
  #pragma unroll
  for (int i = 0; i < 16; ++i) {
    float4 v;
    v.x = __expf(q[i * 4 + 0] - mx) * inv;
    v.y = __expf(q[i * 4 + 1] - mx) * inv;
    v.z = __expf(q[i * 4 + 2] - mx) * inv;
    v.w = __expf(q[i * 4 + 3] - mx) * inv;
    *(float4*)&so[i * 4] = v;
  }
  // Gumbel-max one-hot per 8-category group (argmax invariant to temperature)
  float* st = out + row * 64;
  #pragma unroll
  for (int l = 0; l < 8; ++l) {
    int best = 0; float bv = -3.4e38f;
    #pragma unroll
    for (int c = 0; c < 8; ++c) {
      float u = rng_uniform((unsigned)(row * 64 + l * 8 + c));
      float g = -__logf(-__logf(u + GEPS) + GEPS);
      float v = q[l * 8 + c] + g;
      if (v > bv) { bv = v; best = c; }
    }
    float4 o0, o1;
    o0.x = (best == 0); o0.y = (best == 1); o0.z = (best == 2); o0.w = (best == 3);
    o1.x = (best == 4); o1.y = (best == 5); o1.z = (best == 6); o1.w = (best == 7);
    *(float4*)&st[l * 8 + 0] = o0;
    *(float4*)&st[l * 8 + 4] = o1;
  }
}

// ---------------- host launch ------------------------------------------------
extern "C" void kernel_launch(void* const* d_in, const int* in_sizes, int n_in,
                              void* d_out, int out_size, void* d_ws, size_t ws_size,
                              hipStream_t stream) {
  const float* user_table = (const float*)d_in[0];
  const float* item_table = (const float*)d_in[1];
  const float* rate_table = (const float*)d_in[2];
  const float* ia1_w = (const float*)d_in[3];  const float* ia1_b = (const float*)d_in[4];
  const float* ia2_w = (const float*)d_in[5];  const float* ia2_b = (const float*)d_in[6];
  const float* ia3_w = (const float*)d_in[7];  const float* ia3_b = (const float*)d_in[8];
  const float* ua1_w = (const float*)d_in[9];  const float* ua1_b = (const float*)d_in[10];
  const float* ua2_w = (const float*)d_in[11]; const float* ua2_b = (const float*)d_in[12];
  const float* ua3_w = (const float*)d_in[13]; const float* ua3_b = (const float*)d_in[14];
  const float* rate_w = (const float*)d_in[15]; const float* rate_b = (const float*)d_in[16];
  const float* fuse_w = (const float*)d_in[17]; const float* fuse_b = (const float*)d_in[18];
  const float* self_w = (const float*)d_in[19]; const float* self_b = (const float*)d_in[20];
  const float* exp_w  = (const float*)d_in[21]; const float* exp_b  = (const float*)d_in[22];
  const float* unexp_w = (const float*)d_in[23]; const float* unexp_b = (const float*)d_in[24];
  const float* si1_w = (const float*)d_in[25]; const float* si1_b = (const float*)d_in[26];
  const float* si2_w = (const float*)d_in[27]; const float* si2_b = (const float*)d_in[28];
  const float* um_w  = (const float*)d_in[29]; const float* um_b  = (const float*)d_in[30];
  const float* em_w  = (const float*)d_in[31]; const float* em_b  = (const float*)d_in[32];
  const int* user       = (const int*)d_in[33];
  const int* user_hist  = (const int*)d_in[34];
  const int* user_nbrs  = (const int*)d_in[35];
  const int* unexp_item = (const int*)d_in[36];
  const int* exp_item   = (const int*)d_in[37];

  // workspace layout (half offsets)
  _Float16* w16 = (_Float16*)d_ws;
  _Float16* wRateA = w16 + 0;
  _Float16* wIa1A  = w16 + 4096;
  _Float16* wIa2   = w16 + 8192;
  _Float16* wUa1A  = w16 + 12288;
  _Float16* wUa2   = w16 + 16384;
  _Float16* wFuse  = w16 + 20480;
  _Float16* wSelf  = w16 + 28672;
  _Float16* wExp   = w16 + 36864;
  _Float16* wUnexp = w16 + 40960;
  _Float16* wSi1   = w16 + 45056;
  _Float16* wSi2   = w16 + 49152;
  _Float16* wUm    = w16 + 53248;
  _Float16* wEm    = w16 + 61440;
  _Float16* hi_hs  = w16 + 69632;                       // B x 128
  _Float16* u16    = w16 + 69632 + NBATCH * 128;        // B x 64
  float*    R6     = (float*)(w16 + 69632 + NBATCH * 128 + NBATCH * 64);  // 6 x 64

  // prep: f16 weight conversion (first 64 cols of 128-wide mats where folded)
  conv_kernel<<<16, 256, 0, stream>>>(wRateA, rate_w, 64, 64, 128);
  conv_kernel<<<16, 256, 0, stream>>>(wIa1A, ia1_w, 64, 64, 128);
  conv_kernel<<<16, 256, 0, stream>>>(wIa2, ia2_w, 64, 64, 64);
  conv_kernel<<<16, 256, 0, stream>>>(wUa1A, ua1_w, 64, 64, 128);
  conv_kernel<<<16, 256, 0, stream>>>(wUa2, ua2_w, 64, 64, 64);
  conv_kernel<<<32, 256, 0, stream>>>(wFuse, fuse_w, 64, 128, 128);
  conv_kernel<<<32, 256, 0, stream>>>(wSelf, self_w, 64, 128, 128);
  conv_kernel<<<16, 256, 0, stream>>>(wExp, exp_w, 64, 64, 64);
  conv_kernel<<<16, 256, 0, stream>>>(wUnexp, unexp_w, 64, 64, 64);
  conv_kernel<<<16, 256, 0, stream>>>(wSi1, si1_w, 64, 64, 64);
  conv_kernel<<<16, 256, 0, stream>>>(wSi2, si2_w, 64, 64, 64);
  conv_kernel<<<32, 256, 0, stream>>>(wUm, um_w, 64, 128, 128);
  conv_kernel<<<32, 256, 0, stream>>>(wEm, em_w, 64, 128, 128);
  r6_kernel<<<1, 384, 0, stream>>>(R6, rate_table, rate_w, rate_b);

  attn_kernel<<<NBATCH, 256, 0, stream>>>(
      user_table, item_table, ia1_w, ia1_b, ia2_b, ia3_w, ia3_b,
      ua1_w, ua1_b, ua2_b, ua3_w, ua3_b, user, user_hist, user_nbrs,
      wRateA, wIa1A, wIa2, wUa1A, wUa2, R6, hi_hs, u16);

  head_kernel<<<NBATCH / 64, 256, 0, stream>>>(
      item_table, fuse_b, self_b, exp_b, unexp_b, si1_b, si2_b, um_b, em_b,
      unexp_item, exp_item, wFuse, wSelf, wExp, wUnexp, wSi1, wSi2, wUm, wEm,
      hi_hs, u16, (float*)d_out);

  gumbel_kernel<<<(2 * NBATCH) / 256, 256, 0, stream>>>((float*)d_out);

  (void)in_sizes; (void)n_in; (void)out_size; (void)ws_size;
  (void)ia2_w; (void)ua2_w;
}
